// GEAELayer_33517924778606
// MI455X (gfx1250) — matrix operand order repaired and checked by reading, not compiled, not run on hardware
//
#include <hip/hip_runtime.h>
#include <hip/hip_bf16.h>

// ---------------------------------------------------------------------------
// GEAE layer for MI455X (gfx1250, wave32, WMMA).
// All three GEMMs run on v_wmma_f32_16x16x32_bf16 (bf16 in, f32 accum).
//   k_cvt_bf16  : Wx/We/Wm fp32 -> bf16 (once, RNE) into workspace
//   k_node_proj : xp = x @ Wx + bx          -> bf16 in workspace (L2-resident)
//   k_edge_fused: ef = eattr @ We + be (WMMA -> LDS bf16), then
//                 msg = leaky_relu(concat(xp[src], ef) @ Wm + bm) (WMMA),
//                 scatter-add into d_out via global_atomic_add_f32
//   k_final     : out = sigmoid(out) * relu(beta)   (in place)
// ---------------------------------------------------------------------------

typedef __attribute__((ext_vector_type(16))) __bf16        v16bf;
typedef __attribute__((ext_vector_type(8)))  float         v8f;
typedef __attribute__((ext_vector_type(4)))  float         v4f;
typedef __attribute__((ext_vector_type(4)))  unsigned int  v4u;

union BF16x16 {            // trivial union: 256-bit WMMA operand overlays
  v16bf        v;
  v4u          u[2];
  unsigned int w[8];
};

// float -> bf16, round-to-nearest-even (used only in the one-shot weight cvt)
__device__ __forceinline__ unsigned int f2bf_rne(float f) {
  unsigned int x = __float_as_uint(f);
  return (x + 0x7FFFu + ((x >> 16) & 1u)) >> 16;
}

// pack two floats -> two bf16 (truncation) in ONE v_perm_b32
__device__ __forceinline__ unsigned int pk2t(float lo, float hi) {
  return __builtin_amdgcn_perm(__float_as_uint(hi), __float_as_uint(lo),
                               0x07060302u);
}

// B operand from pre-converted bf16 weights: column gather, p = &W[k0][n];
// elems 0..7 = K [k0..k0+7], elems 8..15 = [k0+16..k0+23] (row stride 128)
__device__ __forceinline__ v16bf load_b_bf(const unsigned short* __restrict__ p) {
  BF16x16 r;
#pragma unroll
  for (int i = 0; i < 8; ++i) {
    const int k = (i < 4) ? 2 * i : 16 + 2 * (i - 4);
    r.w[i] = (unsigned int)p[k * 128] | ((unsigned int)p[(k + 1) * 128] << 16);
  }
  return r.v;
}

// A operand from an fp32 row: elems 0..7 = K [k0..k0+7], elems 8..15 = [k1..k1+7]
__device__ __forceinline__ v16bf load_a_f32(const float* __restrict__ row,
                                            int k0, int k1) {
  const v4f* p0 = (const v4f*)(row + k0);
  const v4f* p1 = (const v4f*)(row + k1);
  v4f a = p0[0], b = p0[1], c = p1[0], d = p1[1];
  BF16x16 r;
  r.w[0] = pk2t(a.x, a.y); r.w[1] = pk2t(a.z, a.w);
  r.w[2] = pk2t(b.x, b.y); r.w[3] = pk2t(b.z, b.w);
  r.w[4] = pk2t(c.x, c.y); r.w[5] = pk2t(c.z, c.w);
  r.w[6] = pk2t(d.x, d.y); r.w[7] = pk2t(d.z, d.w);
  return r.v;
}

__device__ __forceinline__ v8f wmma_bf16(v16bf a, v16bf b, v8f c) {
  return __builtin_amdgcn_wmma_f32_16x16x32_bf16(
      /*neg_a=*/false, a, /*neg_b=*/false, b,
      /*c_mod=*/(short)0, c, /*reuse_a=*/false, /*reuse_b=*/false);
}

// ----------------------- one-shot weight conversion ------------------------
__global__ void k_cvt_bf16(const float* __restrict__ src,
                           unsigned short* __restrict__ dst, int n) {
  int i = blockIdx.x * blockDim.x + threadIdx.x;
  if (i < n) dst[i] = (unsigned short)f2bf_rne(src[i]);
}

// --------------------------- node projection -------------------------------
// xp[N,128] (bf16) = x[N,128] @ Wx[128,128] + bx.  One wave = one 16x16 tile.
__global__ void k_node_proj(const float* __restrict__ x,
                            const unsigned short* __restrict__ Wxb,
                            const float* __restrict__ bx,
                            unsigned short* __restrict__ xp) {
  const int lane  = threadIdx.x & 31;
  const int wave  = blockIdx.x * (blockDim.x >> 5) + (threadIdx.x >> 5);
  const int mtile = wave >> 3;                // 8 column tiles per row tile
  const int nt    = wave & 7;
  const int m     = lane & 15;
  const int off   = (lane & 16) ? 8 : 0;
  const int rbase = (lane & 16) ? 8 : 0;

  const float* arow = x + (size_t)(mtile * 16 + m) * 128;
  v8f acc = {0.f, 0.f, 0.f, 0.f, 0.f, 0.f, 0.f, 0.f};
#pragma unroll
  for (int ks = 0; ks < 4; ++ks) {            // K = 128 = 4 x 32
    const int kb = ks * 32;
    v16bf a = load_a_f32(arow, kb + off, kb + 16 + off);
    v16bf b = load_b_bf(Wxb + (size_t)(kb + off) * 128 + nt * 16 + m);
    acc = wmma_bf16(a, b, acc);
  }
  const int   col  = nt * 16 + m;
  const float bias = bx[col];
#pragma unroll
  for (int r = 0; r < 8; ++r) {
    const int row = mtile * 16 + rbase + r;
    xp[(size_t)row * 128 + col] =
        (unsigned short)(__float_as_uint(acc[r] + bias) >> 16);
  }
}

// ----------------------- fused edge kernel ---------------------------------
// One wave owns 16 edges (a full 16x128 output strip).
__global__ void k_edge_fused(const float*     __restrict__ eattr,
                             const long long* __restrict__ eidx,   // [2,E] i64
                             const unsigned short* __restrict__ Web,
                             const float*     __restrict__ be,
                             const unsigned short* __restrict__ Wmb,
                             const float*     __restrict__ bm,
                             const unsigned short* __restrict__ xp,
                             float* __restrict__ agg, int E) {
  __shared__ unsigned short efT[4][16 * 128];   // 4 KB bf16 ef tile per wave

  const int lane  = threadIdx.x & 31;
  const int w     = threadIdx.x >> 5;
  const int mtile = blockIdx.x * 4 + w;
  const int m     = lane & 15;
  const int off   = (lane & 16) ? 8 : 0;
  const int rbase = (lane & 16) ? 8 : 0;
  const int e     = mtile * 16 + m;

  // ---- phase 1: ef = eattr @ We + be  -> LDS (bf16, row-major 16x128) ----
  const float* arow = eattr + (size_t)e * 64;
  v16bf aE[2];                                  // A depends only on kstep
  aE[0] = load_a_f32(arow, 0 + off, 16 + off);
  aE[1] = load_a_f32(arow, 32 + off, 48 + off);
#pragma unroll
  for (int nt = 0; nt < 8; ++nt) {
    v8f acc = {0.f, 0.f, 0.f, 0.f, 0.f, 0.f, 0.f, 0.f};
#pragma unroll
    for (int ks = 0; ks < 2; ++ks) {            // K = 64 = 2 x 32
      v16bf b = load_b_bf(Web + (size_t)(ks * 32 + off) * 128 + nt * 16 + m);
      acc = wmma_bf16(aE[ks], b, acc);
    }
    const int   col  = nt * 16 + m;
    const float bias = be[col];
#pragma unroll
    for (int r = 0; r < 8; ++r)
      efT[w][(rbase + r) * 128 + col] =
          (unsigned short)(__float_as_uint(acc[r] + bias) >> 16);
  }
  __syncthreads();   // all waves take both phases; guarantees LDS visibility

  // ---- phase 2: msg = leaky_relu(concat(xp[src], ef) @ Wm + bm) ----------
  const long long       se   = eidx[e];                 // src for my A-row
  const unsigned short* xrow = xp + (size_t)se * 128;   // bf16, L2-resident
  long long dstr[8];
#pragma unroll
  for (int r = 0; r < 8; ++r)
    dstr[r] = eidx[(size_t)E + mtile * 16 + rbase + r]; // dst per output row

  // hoist all 8 A ksteps into registers (64 VGPRs), reused by all 8 nt tiles
  BF16x16 aK[8];
#pragma unroll
  for (int ks = 0; ks < 4; ++ks) {              // K 0..127: xp[src] gather
    aK[ks].u[0] = *(const v4u*)(xrow + ks * 32 + off);
    aK[ks].u[1] = *(const v4u*)(xrow + ks * 32 + 16 + off);
  }
#pragma unroll
  for (int ks = 4; ks < 8; ++ks) {              // K 128..255: ef tile in LDS
    const unsigned short* l = &efT[w][m * 128 + (ks - 4) * 32];
    aK[ks].u[0] = *(const v4u*)(l + off);
    aK[ks].u[1] = *(const v4u*)(l + 16 + off);
  }

#pragma unroll
  for (int nt = 0; nt < 8; ++nt) {
    v8f acc = {0.f, 0.f, 0.f, 0.f, 0.f, 0.f, 0.f, 0.f};
#pragma unroll
    for (int ks = 0; ks < 8; ++ks) {            // K = 256 = 8 x 32
      v16bf b = load_b_bf(Wmb + (size_t)(ks * 32 + off) * 128 + nt * 16 + m);
      acc = wmma_bf16(aK[ks].v, b, acc);
    }
    const int   col  = nt * 16 + m;
    const float bias = bm[col];
#pragma unroll
    for (int r = 0; r < 8; ++r) {
      float v = acc[r] + bias;
      v = (v > 0.f) ? v : 0.01f * v;            // leaky_relu, slope 0.01
      atomicAdd(agg + (size_t)dstr[r] * 128 + col, v);  // global_atomic_add_f32
    }
  }
}

// ------------------------------- epilogue ----------------------------------
__global__ void k_zero(float* __restrict__ p, int n) {
  int i = blockIdx.x * blockDim.x + threadIdx.x;
  if (i < n) p[i] = 0.f;
}
__global__ void k_final(float* __restrict__ p, const float* __restrict__ beta,
                        int n) {
  int i = blockIdx.x * blockDim.x + threadIdx.x;
  if (i < n) {
    float s = fmaxf(beta[0], 0.f);
    float a = p[i];
    p[i] = s / (1.f + __expf(-a));
  }
}

// ---------------------------------------------------------------------------
extern "C" void kernel_launch(void* const* d_in, const int* in_sizes, int n_in,
                              void* d_out, int out_size, void* d_ws,
                              size_t ws_size, hipStream_t stream) {
  const float*     x     = (const float*)d_in[0];
  const long long* eidx  = (const long long*)d_in[1];   // int64 [2,E]
  const float*     eattr = (const float*)d_in[2];
  const float*     Wx    = (const float*)d_in[3];
  const float*     bx    = (const float*)d_in[4];
  const float*     We    = (const float*)d_in[5];
  const float*     be    = (const float*)d_in[6];
  const float*     Wm    = (const float*)d_in[7];
  const float*     bm    = (const float*)d_in[8];
  const float*     beta  = (const float*)d_in[9];

  constexpr int N = 50000, E = 600000, OUT = 128;
  constexpr int IN_CH = 128, EDGE_DIM = 64;

  float*          out = (float*)d_out;                 // doubles as agg buffer

  // workspace layout: xp bf16 (12.8 MB) | Wx bf16 | We bf16 | Wm bf16
  unsigned short* xp  = (unsigned short*)d_ws;
  unsigned short* Wxb = (unsigned short*)((char*)d_ws + (size_t)N * OUT * 2);
  unsigned short* Web = Wxb + IN_CH * OUT;             // +16384 elems
  unsigned short* Wmb = Web + EDGE_DIM * OUT;          // +8192 elems

  // one-shot weight conversions (tiny) + accumulator zeroing
  k_cvt_bf16<<<(IN_CH * OUT + 255) / 256, 256, 0, stream>>>(Wx, Wxb,
                                                            IN_CH * OUT);
  k_cvt_bf16<<<(EDGE_DIM * OUT + 255) / 256, 256, 0, stream>>>(We, Web,
                                                               EDGE_DIM * OUT);
  k_cvt_bf16<<<(2 * OUT * OUT + 255) / 256, 256, 0, stream>>>(Wm, Wmb,
                                                              2 * OUT * OUT);
  k_zero<<<(N * OUT + 255) / 256, 256, 0, stream>>>(out, N * OUT);

  // xp = x @ Wx + bx  (3125 row tiles x 8 col tiles = 25000 waves)
  k_node_proj<<<(N / 16) * 8 / 4, 128, 0, stream>>>(x, Wxb, bx, xp);

  // fused ef + msg + scatter-add (37500 waves, 4 per block)
  k_edge_fused<<<(E / 16) / 4, 128, 0, stream>>>(eattr, eidx, Web, be, Wmb, bm,
                                                 xp, out, E);

  // out = sigmoid(agg) * relu(beta), in place
  k_final<<<(N * OUT + 255) / 256, 256, 0, stream>>>(out, beta, N * OUT);
}
